// WODE_73547019977230
// MI455X (gfx1250) — matrix operand order repaired and checked by reading, not compiled, not run on hardware
//
#include <hip/hip_runtime.h>
#include <math.h>

typedef __attribute__((ext_vector_type(16))) _Float16 v16h;
typedef __attribute__((ext_vector_type(8)))  _Float16 v8h;
typedef __attribute__((ext_vector_type(8)))  float    v8f;

#define B_DIM 64
#define L_DIM 128
#define P_DIM 64
#define E_DIM 128
#define H_DIM 128
#define MROWS 64          // rows (l-values) per workgroup
#define NTHREADS 256      // 8 waves of 32
#define LDH 136           // padded f16 row stride (272B -> 4-bank shift per row)

// ---- LDS layout (byte offsets) --------------------------------------------
#define OFF_W1T 0                         // 128 x LDH f16  (W1 transposed, N-major)
#define OFF_W2T 34816                     // 128 x LDH f16
#define OFF_A   69632                     // 64 x LDH f16   (GEMM1 A input)
#define OFF_H   87040                     // 64 x LDH f16   (tanh activations)
#define OFF_Z   104448                    // 64 x 128 f32   (state snapshot for reduce)
#define OFF_B1  137216                    // 128 f32
#define OFF_B2  137728                    // 128 f32
#define OFF_HIS 138240                    // 128 f32
#define OFF_EX  138752                    // 128 f32 (exp terms)
#define OFF_PRE 139264                    // 64 f32
#define OFF_WR  139776                    // 64 f32 (per-row weight)
#define OFF_DEN 140032                    // 1 f32
#define SMEM_BYTES 140064

__device__ __forceinline__ v16h load_a_frag(const _Float16* __restrict__ sA,
                                            int mt, int kc, int lane) {
  // 16x32 f16 A fragment: lanes 0-15 hold M=lane, K = {0..7, 16..23}+koff
  const int m    = lane & 15;
  const int koff = (lane >> 4) << 3;  // 0 or 8
  const _Float16* p = sA + (mt * 16 + m) * LDH + kc * 32 + koff;
  v8h lo = *(const v8h*)p;            // K = koff .. koff+7       (16B aligned)
  v8h hi = *(const v8h*)(p + 16);     // K = 16+koff .. 16+koff+7 (16B aligned)
  return __builtin_shufflevector(lo, hi, 0, 1, 2, 3, 4, 5, 6, 7,
                                 8, 9, 10, 11, 12, 13, 14, 15);
}

__device__ __forceinline__ v16h load_b_frag(const _Float16* __restrict__ sWt,
                                            int nt, int kc, int lane) {
  // 32x16 f16 B fragment: col N=lane&15, K = {0..7, 16..23}+kb; sWt is N-major
  const int n  = lane & 15;
  const int kb = (lane >> 4) << 3;    // 0 or 8
  const _Float16* p = sWt + (nt * 16 + n) * LDH + kc * 32 + kb;
  v8h lo = *(const v8h*)p;
  v8h hi = *(const v8h*)(p + 16);
  return __builtin_shufflevector(lo, hi, 0, 1, 2, 3, 4, 5, 6, 7,
                                 8, 9, 10, 11, 12, 13, 14, 15);
}

__device__ __forceinline__ float fast_tanh(float v) {
  v = fminf(fmaxf(v, -15.0f), 15.0f);
  const float e = __expf(2.0f * v);
  return (e - 1.0f) / (e + 1.0f);
}

__global__ void __launch_bounds__(NTHREADS) wode_zero_kernel(float* __restrict__ out, int n) {
  const int i = blockIdx.x * blockDim.x + threadIdx.x;
  if (i < n) out[i] = 0.0f;
}

__global__ void __launch_bounds__(NTHREADS) wode_rk4_kernel(
    const float* __restrict__ x,   const float* __restrict__ his,
    const float* __restrict__ pre, const float* __restrict__ W1,
    const float* __restrict__ b1,  const float* __restrict__ W2,
    const float* __restrict__ b2,  float* __restrict__ out) {
  extern __shared__ char smem[];
  _Float16* sW1t = (_Float16*)(smem + OFF_W1T);
  _Float16* sW2t = (_Float16*)(smem + OFF_W2T);
  _Float16* sA   = (_Float16*)(smem + OFF_A);
  _Float16* sH   = (_Float16*)(smem + OFF_H);
  float* sZ   = (float*)(smem + OFF_Z);
  float* sB1  = (float*)(smem + OFF_B1);
  float* sB2  = (float*)(smem + OFF_B2);
  float* sHis = (float*)(smem + OFF_HIS);
  float* sEx  = (float*)(smem + OFF_EX);
  float* sPre = (float*)(smem + OFF_PRE);
  float* sWr  = (float*)(smem + OFF_WR);
  float* sDen = (float*)(smem + OFF_DEN);

  const int tid  = threadIdx.x;
  const int lane = tid & 31;
  const int wave = tid >> 5;
  const int bidx = blockIdx.x >> 1;              // batch index
  const int l0   = (blockIdx.x & 1) * MROWS;     // first l owned by this WG

  // Stage weights transposed (N-major) in f16; biases / timeslots in f32.
  for (int i = tid; i < E_DIM * H_DIM; i += NTHREADS) {
    const int k = i >> 7;      // row of W (= E for W1, H for W2)
    const int n = i & 127;     // col of W (= H for W1, E for W2)
    sW1t[n * LDH + k] = (_Float16)W1[i];
    sW2t[n * LDH + k] = (_Float16)W2[i];
  }
  for (int i = tid; i < 128; i += NTHREADS) {
    sB1[i] = b1[i]; sB2[i] = b2[i]; sHis[i] = his[i];
  }
  if (tid < P_DIM) sPre[tid] = pre[tid];

  // Load initial state x -> sZ (f32) and sA (f16 GEMM input).
  for (int i = tid; i < MROWS * E_DIM; i += NTHREADS) {
    const int r = i >> 7, e = i & 127;
    const float v = x[((size_t)bidx * L_DIM + (l0 + r)) * E_DIM + e];
    sZ[r * 128 + e] = v;
    sA[r * LDH + e] = (_Float16)v;
  }
  __syncthreads();

  // Wave -> C-tile assignment: mt fixed, 4 consecutive nt per wave.
  const int mt  = wave & 3;
  const int ntB = (wave >> 2) * 4;
  const int cn  = lane & 15;          // N within tile
  const int cm  = (lane >> 4) * 8;    // M base within tile

  // ---- Hoist all weight B-fragments into registers (invariant over steps) --
  v16h bw1[4][4], bw2[4][4];          // [k-chunk][n-tile]: 256 VGPRs
#pragma unroll
  for (int kc = 0; kc < 4; ++kc)
#pragma unroll
    for (int j = 0; j < 4; ++j) {
      bw1[kc][j] = load_b_frag(sW1t, ntB + j, kc, lane);
      bw2[kc][j] = load_b_frag(sW2t, ntB + j, kc, lane);
    }

  // ODE state for this lane's owned (r,e) elements lives in registers.
  float zreg[4][8];
#pragma unroll
  for (int j = 0; j < 4; ++j)
#pragma unroll
    for (int v = 0; v < 8; ++v)
      zreg[j][v] = sZ[(mt * 16 + cm + v) * 128 + (ntB + j) * 16 + cn];

  // his values for this lane's 8 rows (only needed for the p==0 dt).
  float hisv[8];
#pragma unroll
  for (int v = 0; v < 8; ++v) hisv[v] = sHis[l0 + mt * 16 + cm + v];

  float kacc[4][8];                   // RK4 k-accumulator
  const v8f zero8 = {0.f, 0.f, 0.f, 0.f, 0.f, 0.f, 0.f, 0.f};

  for (int p = 0; p < P_DIM; ++p) {
    // dt: per-row only at p==0, uniform afterwards.
    const float pre0 = sPre[0];
    const float dtp  = (p > 0) ? (sPre[p] - sPre[p - 1]) : 0.0f;

    // Softmax-style output weight for this p.
    if (tid < L_DIM) {
      const float diff = fabsf(sHis[tid] - sPre[p]);
      sEx[tid] = __expf(1.0f / (diff + 1e-8f));
    }
    __syncthreads();
    if (wave == 0) {
      float ssum = sEx[lane] + sEx[lane + 32] + sEx[lane + 64] + sEx[lane + 96];
#pragma unroll
      for (int o = 16; o > 0; o >>= 1) ssum += __shfl_down(ssum, o, 32);
      if (lane == 0) sDen[0] = ssum;
    }
    __syncthreads();
    if (tid < MROWS) sWr[tid] = sEx[l0 + tid] / sDen[0];

    for (int s = 0; s < 4; ++s) {
      __syncthreads();  // sA (and sWr on s==0) ready

      // GEMM1: pre-activation = sA @ W1 (A from LDS, B from registers)
      v8f acc[4];
#pragma unroll
      for (int j = 0; j < 4; ++j) acc[j] = zero8;
#pragma unroll
      for (int kc = 0; kc < 4; ++kc) {
        const v16h a = load_a_frag(sA, mt, kc, lane);
#pragma unroll
        for (int j = 0; j < 4; ++j)
          acc[j] = __builtin_amdgcn_wmma_f32_16x16x32_f16(
              false, a, false, bw1[kc][j], (short)0, acc[j], false, false);
      }
      // tanh(+b1) -> sH (f16)
#pragma unroll
      for (int j = 0; j < 4; ++j) {
        const int e = (ntB + j) * 16 + cn;
        const float bb = sB1[e];
#pragma unroll
        for (int v = 0; v < 8; ++v) {
          const int r = mt * 16 + cm + v;
          sH[r * LDH + e] = (_Float16)fast_tanh(acc[j][v] + bb);
        }
      }
      __syncthreads();

      // GEMM2: k = sH @ W2 + b2
#pragma unroll
      for (int j = 0; j < 4; ++j) acc[j] = zero8;
#pragma unroll
      for (int kc = 0; kc < 4; ++kc) {
        const v16h a = load_a_frag(sH, mt, kc, lane);
#pragma unroll
        for (int j = 0; j < 4; ++j)
          acc[j] = __builtin_amdgcn_wmma_f32_16x16x32_f16(
              false, a, false, bw2[kc][j], (short)0, acc[j], false, false);
      }

      // RK4 stage combine (each (r,e) owned by exactly one lane; z in regs).
#pragma unroll
      for (int j = 0; j < 4; ++j) {
        const int e = (ntB + j) * 16 + cn;
        const float bb = sB2[e];
#pragma unroll
        for (int v = 0; v < 8; ++v) {
          const int r  = mt * 16 + cm + v;
          const int ia = r * LDH + e;
          const float kv = acc[j][v] + bb;
          const float d  = (p == 0) ? (pre0 - hisv[v]) : dtp;
          if (s == 0) {
            kacc[j][v] = kv;
            sA[ia] = (_Float16)(zreg[j][v] + 0.5f * d * kv);
          } else if (s == 1) {
            kacc[j][v] += 2.0f * kv;
            sA[ia] = (_Float16)(zreg[j][v] + 0.5f * d * kv);
          } else if (s == 2) {
            kacc[j][v] += 2.0f * kv;
            sA[ia] = (_Float16)(zreg[j][v] + d * kv);
          } else {
            const float zn = zreg[j][v] + (d * (1.0f / 6.0f)) * (kacc[j][v] + kv);
            zreg[j][v] = zn;                  // new state (registers)
            sZ[r * 128 + e] = zn;             // snapshot for cross-row reduce
            sA[ia] = (_Float16)zn;            // next step's GEMM1 input
          }
        }
      }
    }
    __syncthreads();  // sZ complete for this p

    // out[b,p,e] += sum_r w[l0+r,p] * z_new[r,e]  (128 atomics per WG/step)
    if (tid < E_DIM) {
      float sacc = 0.0f;
#pragma unroll 4
      for (int r = 0; r < MROWS; ++r) sacc += sWr[r] * sZ[r * 128 + tid];
      atomicAdd(&out[((size_t)bidx * P_DIM + p) * E_DIM + tid], sacc);
    }
    __syncthreads();
  }
}

extern "C" void kernel_launch(void* const* d_in, const int* in_sizes, int n_in,
                              void* d_out, int out_size, void* d_ws, size_t ws_size,
                              hipStream_t stream) {
  (void)in_sizes; (void)n_in; (void)out_size; (void)d_ws; (void)ws_size;
  const float* x   = (const float*)d_in[0];
  const float* his = (const float*)d_in[1];
  const float* pre = (const float*)d_in[2];
  const float* W1  = (const float*)d_in[3];
  const float* b1  = (const float*)d_in[4];
  const float* W2  = (const float*)d_in[5];
  const float* b2  = (const float*)d_in[6];
  float* out = (float*)d_out;

  const int n_out = B_DIM * P_DIM * E_DIM;
  wode_zero_kernel<<<(n_out + NTHREADS - 1) / NTHREADS, NTHREADS, 0, stream>>>(out, n_out);

  const dim3 grid(B_DIM * (L_DIM / MROWS));   // 128 workgroups
  wode_rk4_kernel<<<grid, NTHREADS, SMEM_BYTES, stream>>>(
      x, his, pre, W1, b1, W2, b2, out);
}
